// Preprocessor_231928234184
// MI455X (gfx1250) — matrix operand order repaired and checked by reading, not compiled, not run on hardware
//
#include <hip/hip_runtime.h>
#include <hip/hip_bf16.h>

typedef __attribute__((ext_vector_type(16))) _Float16 v16h;
typedef __attribute__((ext_vector_type(8)))  _Float16 v8h;
typedef __attribute__((ext_vector_type(8)))  float    v8f;

#define NPIX      65536        // 256*256
#define ST_STRIDE 264          // 256 + 8 halves pad: rows 16B-aligned, bank-conflict-free
#define TILE_HALVES (256 * ST_STRIDE)   // one padded 256x256 f16 tile in LDS

// ---------------------------------------------------------------------------
// WMMA wrapper: D = A(16x32 f16) * B(32x16 f16) + C(16x16 f32)
// ---------------------------------------------------------------------------
__device__ __forceinline__ v8f wmma16x16x32(v16h a, v16h b, v8f c) {
    return __builtin_amdgcn_wmma_f32_16x16x32_f16(
        /*neg_a=*/false, a, /*neg_b=*/false, b,
        /*c_mod=*/(short)0, c, /*reuse_a=*/false, /*reuse_b=*/false);
}

// A-fragment (16x32) from an f16 row-major matrix with given stride (halves).
// Lane layout: lanes 0-15 -> rows 0-15, K = {kb..kb+7, kb+16..kb+23};
//              lanes 16-31 -> same rows, K = {kb+8.., kb+24..}.
__device__ __forceinline__ v16h load_a_f16(const _Float16* A, int stride,
                                           int mbase, int kb, int lane) {
    int r  = mbase + (lane & 15);
    int hi = lane >> 4;
    const _Float16* p = A + r * stride + kb + hi * 8;
    v8h l0 = *(const v8h*)p;          // 16B aligned
    v8h l1 = *(const v8h*)(p + 16);
    v16h a;
#pragma unroll
    for (int i = 0; i < 8; ++i) { a[i] = l0[i]; a[i + 8] = l1[i]; }
    return a;
}

// B-fragment (32x16). Source stores B column-contiguously: element (k, n) is at
// Bt[(nbase+n)*stride + k]  (i.e. Bt = B^T row-major, or B symmetric).
// Lane layout: lanes 0-15 -> K = {kb..kb+7, kb+16..kb+23}; lanes 16-31 -> K = {kb+8.., kb+24..}.
__device__ __forceinline__ v16h load_b_cols(const _Float16* Bt, int stride,
                                            int nbase, int kb, int lane) {
    int c   = nbase + (lane & 15);
    int off = kb + (lane >> 4) * 8;
    const _Float16* p = Bt + c * stride + off;
    v8h l0 = *(const v8h*)p;          // 16B aligned
    v8h l1 = *(const v8h*)(p + 16);
    v16h b;
#pragma unroll
    for (int i = 0; i < 8; ++i) { b[i] = l0[i]; b[i + 8] = l1[i]; }
    return b;
}

// ---------------------------------------------------------------------------
// Build the two symmetric Gaussian kernels (sigma = 1.0 and sigma = 0.5) in f16
// ---------------------------------------------------------------------------
__global__ void build_g_kernel(_Float16* __restrict__ Gh) {
    int idx = blockIdx.x * blockDim.x + threadIdx.x;   // 0 .. 2*65536-1
    int s = idx >> 16;
    int i = (idx >> 8) & 255;
    int j = idx & 255;
    float d = (float)(i - j);
    float inv2s2 = (s == 0) ? 0.5f : 2.0f;             // 1/(2*sigma^2)
    Gh[idx] = (_Float16)expf(-d * d * inv2s2);
}

// ---------------------------------------------------------------------------
// One workgroup per (batch, map):  g = G * (binarize(X) * G), fully fused:
//   phase 0: binarize X -> LDS Msh (f16)           (x read exactly once)
//   phase 1: S = Msh * G, store S^T -> LDS St      (B-frags: G symmetric)
//   phase 2: g = G * S, B-frags from St rows       (raw g -> out, max -> wmax)
// ---------------------------------------------------------------------------
__global__ void __launch_bounds__(256)
splat_kernel(const float* __restrict__ x, const _Float16* __restrict__ Gh,
             float* __restrict__ out, float* __restrict__ wmax) {
    extern __shared__ _Float16 smem[];          // 2 * 132 KB = 264 KB
    _Float16* Msh = smem;                       // binarized mask, [256][ST_STRIDE]
    _Float16* St  = smem + TILE_HALVES;         // S^T,            [256][ST_STRIDE]

    int bm  = blockIdx.x;
    int b   = bm / 3;
    int map = bm % 3;                              // 0:target 1:cost 2:hist
    int ch  = (map == 0) ? 2 : (map == 1) ? 1 : 3; // source channel in x
    const _Float16* G = Gh + (map == 2 ? NPIX : 0);
    const float* X = x + ((size_t)b * 4 + ch) * NPIX;
    float* O = out + ((size_t)b * 5 + (map + 1)) * NPIX;

    int tid  = threadIdx.x;
    int wave = tid >> 5;
    int lane = tid & 31;
    int hi   = lane >> 4;
    int n0l  = lane & 15;

    // ---- Phase 0: binarize the mask once into LDS (f16) --------------------
#pragma unroll 1
    for (int i = 0; i < 32; ++i) {
        int off = i * 2048 + tid * 8;              // 8 contiguous elements
        int row = off >> 8;
        int col = off & 255;
        v8f f = *(const v8f*)(X + off);            // 32B aligned
        v8h h;
#pragma unroll
        for (int j = 0; j < 8; ++j)
            h[j] = (f[j] > 0.0f) ? (_Float16)1.0f : (_Float16)0.0f;
        *(v8h*)(Msh + row * ST_STRIDE + col) = h;
    }
    __syncthreads();

    // ---- Phase 1: S = M * G ; store S^T (f16) into LDS ----------------------
#pragma unroll 1
    for (int mt = wave; mt < 32; mt += 8) {        // 4 macro-tiles of 64x32 per wave
        int mbase = (mt >> 3) * 64;
        int nbase = (mt & 7) * 32;
        v8f acc[4][2];
#pragma unroll
        for (int mi = 0; mi < 4; ++mi)
#pragma unroll
            for (int ni = 0; ni < 2; ++ni) acc[mi][ni] = (v8f){};

#pragma unroll 1
        for (int kb = 0; kb < 256; kb += 32) {
            v16h a[4], bf[2];
#pragma unroll
            for (int mi = 0; mi < 4; ++mi)
                a[mi] = load_a_f16(Msh, ST_STRIDE, mbase + 16 * mi, kb, lane);
#pragma unroll
            for (int ni = 0; ni < 2; ++ni)
                bf[ni] = load_b_cols(G, 256, nbase + 16 * ni, kb, lane);  // G symmetric
#pragma unroll
            for (int mi = 0; mi < 4; ++mi)
#pragma unroll
                for (int ni = 0; ni < 2; ++ni)
                    acc[mi][ni] = wmma16x16x32(a[mi], bf[ni], acc[mi][ni]);
        }

        // transposed f16 store: St[n][m], contiguous 16B per lane per tile
#pragma unroll
        for (int mi = 0; mi < 4; ++mi)
#pragma unroll
            for (int ni = 0; ni < 2; ++ni) {
                int n0 = nbase + 16 * ni + n0l;
                int m0 = mbase + 16 * mi + 8 * hi;
                v8h hx;
#pragma unroll
                for (int r = 0; r < 8; ++r) hx[r] = (_Float16)acc[mi][ni][r];
                *(v8h*)(St + n0 * ST_STRIDE + m0) = hx;
            }
    }
    __syncthreads();

    // ---- Phase 2: g = G * S ; B-frags come from S^T rows in LDS -------------
    float tmax = 0.0f;
#pragma unroll 1
    for (int mt = wave; mt < 32; mt += 8) {
        int mbase = (mt >> 3) * 64;
        int nbase = (mt & 7) * 32;
        v8f acc[4][2];
#pragma unroll
        for (int mi = 0; mi < 4; ++mi)
#pragma unroll
            for (int ni = 0; ni < 2; ++ni) acc[mi][ni] = (v8f){};

#pragma unroll 1
        for (int kb = 0; kb < 256; kb += 32) {
            v16h a[4], bf[2];
#pragma unroll
            for (int mi = 0; mi < 4; ++mi)
                a[mi] = load_a_f16(G, 256, mbase + 16 * mi, kb, lane);
#pragma unroll
            for (int ni = 0; ni < 2; ++ni)
                bf[ni] = load_b_cols(St, ST_STRIDE, nbase + 16 * ni, kb, lane);
#pragma unroll
            for (int mi = 0; mi < 4; ++mi)
#pragma unroll
                for (int ni = 0; ni < 2; ++ni)
                    acc[mi][ni] = wmma16x16x32(a[mi], bf[ni], acc[mi][ni]);
        }

        // raw f32 output + local max
#pragma unroll
        for (int mi = 0; mi < 4; ++mi)
#pragma unroll
            for (int ni = 0; ni < 2; ++ni) {
                int n = nbase + 16 * ni + n0l;
#pragma unroll
                for (int r = 0; r < 8; ++r) {
                    float v = acc[mi][ni][r];
                    tmax = fmaxf(tmax, v);
                    O[(mbase + 16 * mi + 8 * hi + r) * 256 + n] = v;
                }
            }
    }

    // ---- Workgroup max reduction (reuse Msh region of LDS) ------------------
    __syncthreads();
    float* red = (float*)Msh;
    red[tid] = tmax;
    __syncthreads();
#pragma unroll 1
    for (int s = 128; s > 0; s >>= 1) {
        if (tid < s) red[tid] = fmaxf(red[tid], red[tid + s]);
        __syncthreads();
    }
    if (tid == 0) wmax[bm] = red[0];
}

// ---------------------------------------------------------------------------
// Normalize the three maps in place, copy ch0, compute mul = target*cost.
// ---------------------------------------------------------------------------
__global__ void finalize_kernel(const float* __restrict__ x,
                                const float* __restrict__ wmax,
                                float* __restrict__ out) {
    int idx = blockIdx.x * blockDim.x + threadIdx.x;   // 0 .. 64*65536-1
    int b   = idx >> 16;
    int pix = idx & 65535;

    float mt = wmax[b * 3 + 0]; mt = (mt == 0.0f) ? 1.0f : mt;
    float mc = wmax[b * 3 + 1]; mc = (mc == 0.0f) ? 1.0f : mc;
    float mh = wmax[b * 3 + 2]; mh = (mh == 0.0f) ? 1.0f : mh;

    float* ob = out + (size_t)b * 5 * NPIX;
    float t = ob[1 * NPIX + pix] / mt;
    float c = ob[2 * NPIX + pix] / mc;
    float h = ob[3 * NPIX + pix] / mh;

    ob[0 * NPIX + pix] = x[((size_t)b * 4 + 0) * NPIX + pix];
    ob[1 * NPIX + pix] = t;
    ob[2 * NPIX + pix] = c;
    ob[3 * NPIX + pix] = h;
    ob[4 * NPIX + pix] = t * c;
}

// ---------------------------------------------------------------------------
extern "C" void kernel_launch(void* const* d_in, const int* in_sizes, int n_in,
                              void* d_out, int out_size, void* d_ws, size_t ws_size,
                              hipStream_t stream) {
    (void)in_sizes; (void)n_in; (void)out_size; (void)ws_size;

    const float* x = (const float*)d_in[0];
    float* out = (float*)d_out;

    // workspace layout: [0, 256KB) two f16 256x256 Gaussian kernels; then 192 maxes
    _Float16* Gh   = (_Float16*)d_ws;
    float*    wmax = (float*)((char*)d_ws + (size_t)2 * NPIX * sizeof(_Float16));

    const int smem = 2 * TILE_HALVES * (int)sizeof(_Float16);   // 270336 B
    (void)hipFuncSetAttribute((const void*)splat_kernel,
                              hipFuncAttributeMaxDynamicSharedMemorySize, smem);

    build_g_kernel<<<(2 * NPIX) / 256, 256, 0, stream>>>(Gh);
    splat_kernel<<<192, 256, smem, stream>>>(x, Gh, out, wmax);
    finalize_kernel<<<(64 * NPIX) / 256, 256, 0, stream>>>(x, wmax, out);
}
